// WorldModelV3A1Full_59923383714613
// MI455X (gfx1250) — compile-verified, matrix operand order checked by reading, not run on hardware
//
#include <hip/hip_runtime.h>
#include <hip/hip_bf16.h>

// ---------------------------------------------------------------------------
// WorldModelV3A1Full forward for MI455X (gfx1250), wave32 + WMMA f16->f32.
//
// GEMMs: f32 I/O, f16 operands, f32 accumulate via v_wmma_f32_16x16x32_f16.
// Register-blocked over N (NT accumulators per wave; NT=6 covers D=96 in one
// wave, A fragment reused NT times per K step). All fragment gathers are
// BRANCHLESS: M/N are multiples of 16 everywhere so no row/col guards; K tails
// (K=24) use clamped addresses + value selects (v_cndmask) instead of
// exec-masked per-element loads. Attention is fused (scores -> softmax -> P@V)
// per 16-row tile in LDS. LN / softmax / LSTM pointwise stay f32.
// ---------------------------------------------------------------------------

#define B_   512
#define T_   256
#define F_   24
#define D_   96
#define NH_  4
#define DH_  24
#define DFF_ 384
#define NL_  3
#define HID_ 96
#define HOR_ 40

typedef __attribute__((ext_vector_type(16))) _Float16 v16h;
typedef __attribute__((ext_vector_type(8)))  float    v8f;

#define GF_BIAS   1
#define GF_RELU   2
#define GF_ACCUM  4
#define GF_RESID  8
#define GF_TRANSB 16

__device__ __forceinline__ float sigmoidf_(float x) { return 1.f / (1.f + __expf(-x)); }

__device__ __forceinline__ v8f wmma_f16(v16h a, v16h b, v8f c) {
  // (neg_a, A, neg_b, B, c_mod, C, reuse_a, reuse_b)
  return __builtin_amdgcn_wmma_f32_16x16x32_f16(false, a, false, b, (short)0, c, false, false);
}

// A fragment: 16x32 f16, row = lane%16, K split per CDNA5 ISA layout:
//  lanes 0-15: k = {k0..k0+7} then {k0+16..k0+23}; lanes 16-31: +8 on each.
// Requires row valid (M % 16 == 0). K tail handled branchlessly.
__device__ __forceinline__ v16h load_a_frag(const float* __restrict__ A, int lda,
                                            int row, int k0, int K) {
  int lane = threadIdx.x & 31;
  int koff = (lane < 16) ? 0 : 8;
  const float* rp = A + (size_t)row * lda;
  v16h a;
  if (k0 + 32 <= K) {
    float buf[16];
    const float4* p0 = reinterpret_cast<const float4*>(rp + k0 + koff);
    const float4* p1 = reinterpret_cast<const float4*>(rp + k0 + 16 + koff);
    *reinterpret_cast<float4*>(&buf[0])  = p0[0];
    *reinterpret_cast<float4*>(&buf[4])  = p0[1];
    *reinterpret_cast<float4*>(&buf[8])  = p1[0];
    *reinterpret_cast<float4*>(&buf[12]) = p1[1];
#pragma unroll
    for (int i = 0; i < 8; ++i) { a[i] = (_Float16)buf[i]; a[i + 8] = (_Float16)buf[8 + i]; }
  } else {
    // branchless tail: clamp address (always valid), select value to 0
#pragma unroll
    for (int i = 0; i < 8; ++i) {
      int ka = k0 + koff + i, kb = k0 + 16 + koff + i;
      int kaC = (ka < K) ? ka : (K - 1);
      int kbC = (kb < K) ? kb : (K - 1);
      float ta = rp[kaC];
      float tb = rp[kbC];
      float fa = (ka < K) ? ta : 0.f;
      float fb = (kb < K) ? tb : 0.f;
      a[i] = (_Float16)fa; a[i + 8] = (_Float16)fb;
    }
  }
  return a;
}

// B fragment: 32x16 f16, col = lane%16, same K split. transB => B[k][n]=Bsrc[n*ldb+k]
// Requires col valid (N % 16 == 0). K tail handled branchlessly.
__device__ __forceinline__ v16h load_b_frag(const float* __restrict__ Bm, int ldb,
                                            int k0, int col, int K, bool transB) {
  int lane = threadIdx.x & 31;
  int koff = (lane < 16) ? 0 : 8;
  v16h b;
  if (transB) {
    const float* cp = Bm + (size_t)col * ldb;
    if (k0 + 32 <= K) {
      float buf[16];
      const float4* p0 = reinterpret_cast<const float4*>(cp + k0 + koff);
      const float4* p1 = reinterpret_cast<const float4*>(cp + k0 + 16 + koff);
      *reinterpret_cast<float4*>(&buf[0])  = p0[0];
      *reinterpret_cast<float4*>(&buf[4])  = p0[1];
      *reinterpret_cast<float4*>(&buf[8])  = p1[0];
      *reinterpret_cast<float4*>(&buf[12]) = p1[1];
#pragma unroll
      for (int i = 0; i < 8; ++i) { b[i] = (_Float16)buf[i]; b[i + 8] = (_Float16)buf[8 + i]; }
    } else {
#pragma unroll
      for (int i = 0; i < 8; ++i) {
        int ka = k0 + koff + i, kb = k0 + 16 + koff + i;
        int kaC = (ka < K) ? ka : (K - 1);
        int kbC = (kb < K) ? kb : (K - 1);
        float ta = cp[kaC];
        float tb = cp[kbC];
        float fa = (ka < K) ? ta : 0.f;
        float fb = (kb < K) ? tb : 0.f;
        b[i] = (_Float16)fa; b[i + 8] = (_Float16)fb;
      }
    }
  } else {
    if (k0 + 32 <= K) {
#pragma unroll
      for (int i = 0; i < 8; ++i) {
        int ka = k0 + koff + i, kb = k0 + 16 + koff + i;
        float fa = Bm[(size_t)ka * ldb + col];
        float fb = Bm[(size_t)kb * ldb + col];
        b[i] = (_Float16)fa; b[i + 8] = (_Float16)fb;
      }
    } else {
#pragma unroll
      for (int i = 0; i < 8; ++i) {
        int ka = k0 + koff + i, kb = k0 + 16 + koff + i;
        int kaC = (ka < K) ? ka : (K - 1);
        int kbC = (kb < K) ? kb : (K - 1);
        float ta = Bm[(size_t)kaC * ldb + col];
        float tb = Bm[(size_t)kbC * ldb + col];
        float fa = (ka < K) ? ta : 0.f;
        float fb = (kb < K) ? tb : 0.f;
        b[i] = (_Float16)fa; b[i + 8] = (_Float16)fb;
      }
    }
  }
  return b;
}

// Generic WMMA GEMM, register-blocked over N: each wave owns NT 16x16 tiles in
// one row-tile and reuses its A fragment NT times per K step.
// C[M,N] = ep(A[M,K] @ op(B) + bias) (+C) (+resid).
// Requires M % 16 == 0, N % 16 == 0, tilesN % NT == 0.
template <int NT>
__global__ void __launch_bounds__(128)
k_wmma_gemm_t(const float* __restrict__ A, int lda,
              const float* __restrict__ Bm, int ldb,
              const float* __restrict__ bias,
              const float* __restrict__ resid,
              float* __restrict__ C, int ldc,
              int M, int N, int K, int flags) {
  int tilesN = N >> 4;
  int groupsN = tilesN / NT;
  int tilesM = M >> 4;
  int wave = blockIdx.x * 4 + (threadIdx.x >> 5);
  if (wave >= tilesM * groupsN) return;          // wave-uniform: EXEC stays full
  int tm = wave / groupsN, tg = wave - tm * groupsN;
  int tn0 = tg * NT;
  int lane = threadIdx.x & 31;
  int arow = tm * 16 + (lane & 15);
  bool transB = (flags & GF_TRANSB) != 0;

  v8f acc[NT];
#pragma unroll
  for (int j = 0; j < NT; ++j) acc[j] = (v8f){0.f, 0.f, 0.f, 0.f, 0.f, 0.f, 0.f, 0.f};

  for (int k0 = 0; k0 < K; k0 += 32) {
    if (k0 + 32 < K)  // next-tile prefetch -> global_prefetch_b8
      __builtin_prefetch(A + (size_t)arow * lda + k0 + 32, 0, 1);
    v16h a = load_a_frag(A, lda, arow, k0, K);
#pragma unroll
    for (int j = 0; j < NT; ++j) {
      int bcol = (tn0 + j) * 16 + (lane & 15);
      v16h b = load_b_frag(Bm, ldb, k0, bcol, K, transB);
      acc[j] = wmma_f16(a, b, acc[j]);
    }
  }

  // D layout: VGPR i -> row i (lanes 0-15) / row i+8 (lanes 16-31), col = lane%16
  int rowoff = (lane < 16) ? 0 : 8;
#pragma unroll
  for (int j = 0; j < NT; ++j) {
    int col = (tn0 + j) * 16 + (lane & 15);
#pragma unroll
    for (int i = 0; i < 8; ++i) {
      int row = tm * 16 + rowoff + i;
      float v = acc[j][i];
      if (flags & GF_BIAS) v += bias[col];
      if (flags & GF_RELU) v = fmaxf(v, 0.f);
      size_t idx = (size_t)row * ldc + col;
      if (flags & GF_ACCUM) v += C[idx];
      if (flags & GF_RESID) v += resid[idx];
      C[idx] = v;
    }
  }
}

// Fused attention: per (b, h, 16-row tile): scores (WMMA) -> softmax (LDS) -> P@V (WMMA)
// DH=24 is zero-padded to K=32: fragment elements 0..7 (k = koff+i <= 15) never
// need masking; elements 8..15 (k = 16+koff+i) are masked only for lanes 16-31,
// done branchlessly with a clamped load + select.
#define PSTRIDE 260  // 260 % 64 == 4 -> conflict-free column sweeps in softmax
extern "C" __global__ void __launch_bounds__(64)
k_attention(const float* __restrict__ q, const float* __restrict__ kM,
            const float* __restrict__ vM, float* __restrict__ o) {
  __shared__ float sp[2][16][PSTRIDE];  // 33 KB
  int w = threadIdx.x >> 5;
  int lane = threadIdx.x & 31;
  int gw = blockIdx.x * 2 + w;       // B*NH*(T/16) = 32768 waves, exact grid
  int rt = gw & 15;
  int h  = (gw >> 4) & (NH_ - 1);
  int b  = gw >> 6;
  int koff = (lane < 16) ? 0 : 8;
  const float scale = 0.20412414523193154f;  // 1/sqrt(24)

  // A fragment (q rows) -- loaded once
  int qrow = rt * 16 + (lane & 15);
  const float* qp = q + ((size_t)(b * T_ + qrow)) * D_ + h * DH_;
  v16h qa;
#pragma unroll
  for (int i = 0; i < 8; ++i) {
    int kb = 16 + koff + i;
    int kbC = (kb < DH_) ? kb : (DH_ - 1);
    float fa = qp[koff + i];                  // always valid (k <= 15)
    float tb = qp[kbC];
    float fb = (kb < DH_) ? tb : 0.f;
    qa[i] = (_Float16)fa; qa[i + 8] = (_Float16)fb;
  }

  // scores: 16 column tiles of 16
  for (int tn = 0; tn < 16; ++tn) {
    int srow = tn * 16 + (lane & 15);
    const float* kp = kM + ((size_t)(b * T_ + srow)) * D_ + h * DH_;
    v16h kf;
#pragma unroll
    for (int i = 0; i < 8; ++i) {
      int kb = 16 + koff + i;
      int kbC = (kb < DH_) ? kb : (DH_ - 1);
      float fa = kp[koff + i];
      float tb = kp[kbC];
      float fb = (kb < DH_) ? tb : 0.f;
      kf[i] = (_Float16)fa; kf[i + 8] = (_Float16)fb;
    }
    v8f acc = {0.f, 0.f, 0.f, 0.f, 0.f, 0.f, 0.f, 0.f};
    acc = wmma_f16(qa, kf, acc);
    int rowoff = (lane < 16) ? 0 : 8;
    int cc = tn * 16 + (lane & 15);
#pragma unroll
    for (int i = 0; i < 8; ++i) sp[w][rowoff + i][cc] = acc[i] * scale;
  }
  __syncthreads();

  // softmax: two lanes per row (split 128/128), combine via shfl_xor(16)
  {
    int r = lane & 15;
    int j0 = (lane >> 4) * 128;
    float m = -3.4e38f;
    for (int j = 0; j < 128; ++j) m = fmaxf(m, sp[w][r][j0 + j]);
    m = fmaxf(m, __shfl_xor(m, 16));
    float s = 0.f;
    for (int j = 0; j < 128; ++j) {
      float e = __expf(sp[w][r][j0 + j] - m);
      sp[w][r][j0 + j] = e; s += e;
    }
    s += __shfl_xor(s, 16);
    float inv = 1.f / s;
    for (int j = 0; j < 128; ++j) sp[w][r][j0 + j] *= inv;
  }
  __syncthreads();

  // o = P[16,256] @ V[256,24] : K in 8 chunks of 32, N in two 16-col tiles (24 valid)
  v8f oa0 = {0.f, 0.f, 0.f, 0.f, 0.f, 0.f, 0.f, 0.f};
  v8f oa1 = {0.f, 0.f, 0.f, 0.f, 0.f, 0.f, 0.f, 0.f};
  int prow = lane & 15;
  int vc1 = 16 + (lane & 15);
  int vc1C = (vc1 < DH_) ? vc1 : (DH_ - 1);
  bool c1OK = vc1 < DH_;
  for (int kt = 0; kt < 8; ++kt) {
    v16h pf;
#pragma unroll
    for (int i = 0; i < 8; ++i) {
      pf[i]     = (_Float16)sp[w][prow][kt * 32 + koff + i];
      pf[i + 8] = (_Float16)sp[w][prow][kt * 32 + 16 + koff + i];
    }
    {  // nt = 0 : cols 0..15 (always valid)
      int vc = lane & 15;
      v16h vf;
#pragma unroll
      for (int i = 0; i < 8; ++i) {
        vf[i]     = (_Float16)vM[((size_t)(b * T_) + kt * 32 + koff + i) * D_ + h * DH_ + vc];
        vf[i + 8] = (_Float16)vM[((size_t)(b * T_) + kt * 32 + 16 + koff + i) * D_ + h * DH_ + vc];
      }
      oa0 = wmma_f16(pf, vf, oa0);
    }
    {  // nt = 1 : cols 16..23, branchless column mask
      v16h vf;
#pragma unroll
      for (int i = 0; i < 8; ++i) {
        float ta = vM[((size_t)(b * T_) + kt * 32 + koff + i) * D_ + h * DH_ + vc1C];
        float tb = vM[((size_t)(b * T_) + kt * 32 + 16 + koff + i) * D_ + h * DH_ + vc1C];
        float fa = c1OK ? ta : 0.f;
        float fb = c1OK ? tb : 0.f;
        vf[i] = (_Float16)fa; vf[i + 8] = (_Float16)fb;
      }
      oa1 = wmma_f16(pf, vf, oa1);
    }
  }
  int rowoff = (lane < 16) ? 0 : 8;
  int col0 = lane & 15;
#pragma unroll
  for (int i = 0; i < 8; ++i)
    o[((size_t)(b * T_) + rt * 16 + rowoff + i) * D_ + h * DH_ + col0] = oa0[i];
  if (c1OK) {
#pragma unroll
    for (int i = 0; i < 8; ++i)
      o[((size_t)(b * T_) + rt * 16 + rowoff + i) * D_ + h * DH_ + vc1] = oa1[i];
  }
}

// x += positional encoding
extern "C" __global__ void k_posenc(float* __restrict__ x) {
  size_t idx = (size_t)blockIdx.x * blockDim.x + threadIdx.x;
  if (idx >= (size_t)B_ * T_ * D_) return;
  int d = (int)(idx % D_);
  int t = (int)((idx / D_) % T_);
  int i2 = (d >> 1) << 1;
  float div = __expf((float)i2 * (-9.210340371976184f / (float)D_));
  float ang = (float)t * div;
  x[idx] += (d & 1) ? __cosf(ang) : __sinf(ang);
}

// LayerNorm over width 96: one wave per row (3 elems/lane, shfl_xor reduce)
extern "C" __global__ void __launch_bounds__(128)
k_layernorm(const float* __restrict__ in, const float* __restrict__ g,
            const float* __restrict__ bb, float* __restrict__ out, int rows) {
  int row = blockIdx.x * 4 + (threadIdx.x >> 5);
  if (row >= rows) return;
  int lane = threadIdx.x & 31;
  const float* r = in + (size_t)row * D_;
  float x0 = r[lane], x1 = r[lane + 32], x2 = r[lane + 64];
  float s = x0 + x1 + x2;
#pragma unroll
  for (int o = 16; o; o >>= 1) s += __shfl_xor(s, o);
  float mu = s * (1.f / D_);
  float d0 = x0 - mu, d1 = x1 - mu, d2 = x2 - mu;
  float v = d0 * d0 + d1 * d1 + d2 * d2;
#pragma unroll
  for (int o = 16; o; o >>= 1) v += __shfl_xor(v, o);
  float inv = rsqrtf(v * (1.f / D_) + 1e-5f);
  float* wout = out + (size_t)row * D_;
  wout[lane]      = d0 * inv * g[lane]      + bb[lane];
  wout[lane + 32] = d1 * inv * g[lane + 32] + bb[lane + 32];
  wout[lane + 64] = d2 * inv * g[lane + 64] + bb[lane + 64];
}

// mean over T
extern "C" __global__ void k_mean(const float* __restrict__ x, float* __restrict__ emb) {
  int idx = blockIdx.x * blockDim.x + threadIdx.x;
  if (idx >= B_ * D_) return;
  int b = idx / D_, d = idx % D_;
  const float* p = x + (size_t)b * T_ * D_ + d;
  float s = 0.f;
  for (int t = 0; t < T_; ++t) s += p[(size_t)t * D_];
  emb[idx] = s * (1.f / T_);
}

// eol / hi heads: one wave per batch row
extern "C" __global__ void __launch_bounds__(128)
k_heads(const float* __restrict__ hs,
        const float* __restrict__ rul_W, const float* __restrict__ rul_b,
        const float* __restrict__ hi_W,  const float* __restrict__ hi_b,
        float* __restrict__ out_eol, float* __restrict__ out_hi, float* __restrict__ inp) {
  int b = blockIdx.x * 4 + (threadIdx.x >> 5);
  if (b >= B_) return;
  int lane = threadIdx.x & 31;
  const float* hr = hs + (size_t)b * D_;
  float x0 = hr[lane], x1 = hr[lane + 32], x2 = hr[lane + 64];
  float dr = x0 * rul_W[lane] + x1 * rul_W[lane + 32] + x2 * rul_W[lane + 64];
  float dh = x0 * hi_W[lane]  + x1 * hi_W[lane + 32]  + x2 * hi_W[lane + 64];
#pragma unroll
  for (int o = 16; o; o >>= 1) { dr += __shfl_xor(dr, o); dh += __shfl_xor(dh, o); }
  if (lane == 0) {
    out_eol[b] = dr + rul_b[0];
    float hv = sigmoidf_(dh + hi_b[0]);
    out_hi[b] = hv;
    inp[b] = hv;
  }
}

extern "C" __global__ void k_init_states(const float* __restrict__ e2h, const float* __restrict__ e2c,
                                         float* __restrict__ h0, float* __restrict__ h1,
                                         float* __restrict__ c0, float* __restrict__ c1) {
  int idx = blockIdx.x * blockDim.x + threadIdx.x;
  if (idx >= B_ * HID_) return;
  int b = idx / HID_, j = idx % HID_;
  h0[idx] = e2h[(size_t)b * 2 * HID_ + j];
  h1[idx] = e2h[(size_t)b * 2 * HID_ + HID_ + j];
  c0[idx] = e2c[(size_t)b * 2 * HID_ + j];
  c1[idx] = e2c[(size_t)b * 2 * HID_ + HID_ + j];
}

// LSTM pointwise cell: gates (i,f,g,o) from G [B,4H] (+ rank-1 inp*Wih + biases)
extern "C" __global__ void k_lstm_cell(const float* __restrict__ G,
                                       const float* __restrict__ wih_col,
                                       const float* __restrict__ inp,
                                       const float* __restrict__ bih_l,
                                       const float* __restrict__ bhh_l,
                                       float* __restrict__ h, float* __restrict__ c) {
  int idx = blockIdx.x * blockDim.x + threadIdx.x;
  if (idx >= B_ * HID_) return;
  int b = idx / HID_, j = idx % HID_;
  const float* g = G + (size_t)b * 4 * HID_;
  float gi = g[j]             + bih_l[j]             + bhh_l[j];
  float gf = g[HID_ + j]      + bih_l[HID_ + j]      + bhh_l[HID_ + j];
  float gg = g[2 * HID_ + j]  + bih_l[2 * HID_ + j]  + bhh_l[2 * HID_ + j];
  float go = g[3 * HID_ + j]  + bih_l[3 * HID_ + j]  + bhh_l[3 * HID_ + j];
  if (wih_col != nullptr) {
    float add = inp[b];
    gi += add * wih_col[j];
    gf += add * wih_col[HID_ + j];
    gg += add * wih_col[2 * HID_ + j];
    go += add * wih_col[3 * HID_ + j];
  }
  float cn = sigmoidf_(gf) * c[idx] + sigmoidf_(gi) * tanhf(gg);
  c[idx] = cn;
  h[idx] = sigmoidf_(go) * tanhf(cn);
}

// traj head: pred = h1 @ traj_W + b, written to output and fed back as next input
extern "C" __global__ void __launch_bounds__(128)
k_traj(const float* __restrict__ h1, const float* __restrict__ W, const float* __restrict__ bias,
       float* __restrict__ out_traj, float* __restrict__ inp, int t) {
  int b = blockIdx.x * 4 + (threadIdx.x >> 5);
  if (b >= B_) return;
  int lane = threadIdx.x & 31;
  const float* hr = h1 + (size_t)b * HID_;
  float s = hr[lane] * W[lane] + hr[lane + 32] * W[lane + 32] + hr[lane + 64] * W[lane + 64];
#pragma unroll
  for (int o = 16; o; o >>= 1) s += __shfl_xor(s, o);
  if (lane == 0) {
    float p = s + bias[0];
    out_traj[(size_t)b * HOR_ + t] = p;
    inp[b] = p;
  }
}

// ---------------------------------------------------------------------------

static void gemm(hipStream_t s, const float* A, int lda, const float* Bm, int ldb,
                 const float* bias, const float* resid, float* C, int ldc,
                 int M, int N, int K, int flags) {
  int tilesM = M / 16, tilesN = N / 16;
  if (tilesN % 6 == 0) {
    int waves = tilesM * (tilesN / 6);
    k_wmma_gemm_t<6><<<(waves + 3) / 4, 128, 0, s>>>(A, lda, Bm, ldb, bias, resid, C, ldc, M, N, K, flags);
  } else if (tilesN % 4 == 0) {
    int waves = tilesM * (tilesN / 4);
    k_wmma_gemm_t<4><<<(waves + 3) / 4, 128, 0, s>>>(A, lda, Bm, ldb, bias, resid, C, ldc, M, N, K, flags);
  } else if (tilesN % 2 == 0) {
    int waves = tilesM * (tilesN / 2);
    k_wmma_gemm_t<2><<<(waves + 3) / 4, 128, 0, s>>>(A, lda, Bm, ldb, bias, resid, C, ldc, M, N, K, flags);
  } else {
    int waves = tilesM * tilesN;
    k_wmma_gemm_t<1><<<(waves + 3) / 4, 128, 0, s>>>(A, lda, Bm, ldb, bias, resid, C, ldc, M, N, K, flags);
  }
}

extern "C" void kernel_launch(void* const* d_in, const int* in_sizes, int n_in,
                              void* d_out, int out_size, void* d_ws, size_t ws_size,
                              hipStream_t stream) {
  (void)in_sizes; (void)n_in; (void)out_size; (void)ws_size;
  // setup_inputs dict order: encoder_inputs, then params in insertion order
  const float* enc      = (const float*)d_in[0];
  const float* proj_W   = (const float*)d_in[1];
  const float* proj_b   = (const float*)d_in[2];
  const float* Wq       = (const float*)d_in[3];
  const float* Wq_b     = (const float*)d_in[4];
  const float* Wk       = (const float*)d_in[5];
  const float* Wk_b     = (const float*)d_in[6];
  const float* Wv       = (const float*)d_in[7];
  const float* Wv_b     = (const float*)d_in[8];
  const float* Wo       = (const float*)d_in[9];
  const float* Wo_b     = (const float*)d_in[10];
  const float* W1       = (const float*)d_in[11];
  const float* b1       = (const float*)d_in[12];
  const float* W2       = (const float*)d_in[13];
  const float* b2       = (const float*)d_in[14];
  const float* ln1_g    = (const float*)d_in[15];
  const float* ln1_b    = (const float*)d_in[16];
  const float* ln2_g    = (const float*)d_in[17];
  const float* ln2_b    = (const float*)d_in[18];
  const float* shared_W = (const float*)d_in[19];
  const float* shared_b = (const float*)d_in[20];
  const float* rul_W    = (const float*)d_in[21];
  const float* rul_b    = (const float*)d_in[22];
  const float* hi_W     = (const float*)d_in[23];
  const float* hi_b     = (const float*)d_in[24];
  const float* e2h_W    = (const float*)d_in[25];
  const float* e2h_b    = (const float*)d_in[26];
  const float* e2c_W    = (const float*)d_in[27];
  const float* e2c_b    = (const float*)d_in[28];
  const float* Wih0     = (const float*)d_in[29];
  const float* Whh0     = (const float*)d_in[30];
  const float* Wih1     = (const float*)d_in[31];
  const float* Whh1     = (const float*)d_in[32];
  const float* bih      = (const float*)d_in[33];
  const float* bhh      = (const float*)d_in[34];
  const float* traj_W   = (const float*)d_in[35];
  const float* traj_b   = (const float*)d_in[36];

  float* ws = (float*)d_ws;
  size_t off = 0;
  auto take = [&](size_t n) { float* p = ws + off; off += n; return p; };
  const size_t BT = (size_t)B_ * T_;
  float* x   = take(BT * D_);
  float* t1  = take(BT * D_);
  float* qb  = take(BT * D_);
  float* kb  = take(BT * D_);
  float* vb  = take(BT * D_);
  float* ob  = take(BT * D_);
  float* ff  = take(BT * DFF_);
  float* emb = take((size_t)B_ * D_);
  float* hsh = take((size_t)B_ * D_);
  float* e2h = take((size_t)B_ * 2 * HID_);
  float* e2c = take((size_t)B_ * 2 * HID_);
  float* h0  = take((size_t)B_ * HID_);
  float* h1s = take((size_t)B_ * HID_);
  float* c0  = take((size_t)B_ * HID_);
  float* c1s = take((size_t)B_ * HID_);
  float* G0  = take((size_t)B_ * 4 * HID_);
  float* G1  = take((size_t)B_ * 4 * HID_);
  float* inp = take((size_t)B_);

  float* out_traj = (float*)d_out;                       // [B, HOR]
  float* out_eol  = out_traj + (size_t)B_ * HOR_;        // [B]
  float* out_hi   = out_eol + B_;                        // [B]

  // input projection + positional encoding
  gemm(stream, enc, F_, proj_W, D_, proj_b, nullptr, x, D_, (int)BT, D_, F_, GF_BIAS);
  {
    size_t n = BT * D_;
    k_posenc<<<(unsigned)((n + 255) / 256), 256, 0, stream>>>(x);
  }

  // transformer layers
  for (int l = 0; l < NL_; ++l) {
    const float* wq = Wq + (size_t)l * D_ * D_;
    const float* wk = Wk + (size_t)l * D_ * D_;
    const float* wv = Wv + (size_t)l * D_ * D_;
    const float* wo = Wo + (size_t)l * D_ * D_;
    gemm(stream, x, D_, wq, D_, Wq_b + l * D_, nullptr, qb, D_, (int)BT, D_, D_, GF_BIAS);
    gemm(stream, x, D_, wk, D_, Wk_b + l * D_, nullptr, kb, D_, (int)BT, D_, D_, GF_BIAS);
    gemm(stream, x, D_, wv, D_, Wv_b + l * D_, nullptr, vb, D_, (int)BT, D_, D_, GF_BIAS);
    k_attention<<<16384, 64, 0, stream>>>(qb, kb, vb, ob);
    gemm(stream, ob, D_, wo, D_, Wo_b + l * D_, x, t1, D_, (int)BT, D_, D_, GF_BIAS | GF_RESID);
    k_layernorm<<<(unsigned)(BT / 4), 128, 0, stream>>>(t1, ln1_g + l * D_, ln1_b + l * D_, x, (int)BT);
    gemm(stream, x, D_, W1 + (size_t)l * D_ * DFF_, DFF_, b1 + l * DFF_, nullptr, ff, DFF_,
         (int)BT, DFF_, D_, GF_BIAS | GF_RELU);
    gemm(stream, ff, DFF_, W2 + (size_t)l * DFF_ * D_, D_, b2 + l * D_, x, t1, D_,
         (int)BT, D_, DFF_, GF_BIAS | GF_RESID);
    k_layernorm<<<(unsigned)(BT / 4), 128, 0, stream>>>(t1, ln2_g + l * D_, ln2_b + l * D_, x, (int)BT);
  }

  // pooled embedding + heads
  k_mean<<<(B_ * D_ + 255) / 256, 256, 0, stream>>>(x, emb);
  gemm(stream, emb, D_, shared_W, D_, shared_b, nullptr, hsh, D_, B_, D_, D_, GF_BIAS | GF_RELU);
  k_heads<<<B_ / 4, 128, 0, stream>>>(hsh, rul_W, rul_b, hi_W, hi_b, out_eol, out_hi, inp);
  gemm(stream, emb, D_, e2h_W, 2 * HID_, e2h_b, nullptr, e2h, 2 * HID_, B_, 2 * HID_, D_, GF_BIAS);
  gemm(stream, emb, D_, e2c_W, 2 * HID_, e2c_b, nullptr, e2c, 2 * HID_, B_, 2 * HID_, D_, GF_BIAS);
  k_init_states<<<(B_ * HID_ + 255) / 256, 256, 0, stream>>>(e2h, e2c, h0, h1s, c0, c1s);

  // LSTM rollout (sequential); Whh/Wih are [4H, H] so B is transposed (ldb = H)
  for (int t = 0; t < HOR_; ++t) {
    gemm(stream, h0, HID_, Whh0, HID_, nullptr, nullptr, G0, 4 * HID_,
         B_, 4 * HID_, HID_, GF_TRANSB);
    k_lstm_cell<<<(B_ * HID_ + 255) / 256, 256, 0, stream>>>(G0, Wih0, inp, bih, bhh, h0, c0);
    gemm(stream, h0, HID_, Wih1, HID_, nullptr, nullptr, G1, 4 * HID_,
         B_, 4 * HID_, HID_, GF_TRANSB);
    gemm(stream, h1s, HID_, Whh1, HID_, nullptr, nullptr, G1, 4 * HID_,
         B_, 4 * HID_, HID_, GF_TRANSB | GF_ACCUM);
    k_lstm_cell<<<(B_ * HID_ + 255) / 256, 256, 0, stream>>>(G1, nullptr, nullptr,
                                                             bih + 4 * HID_, bhh + 4 * HID_, h1s, c1s);
    k_traj<<<B_ / 4, 128, 0, stream>>>(h1s, traj_W, traj_b, out_traj, inp, t);
  }
}
